// FixedMemoryModule_49203145343273
// MI455X (gfx1250) — matrix operand order repaired and checked by reading, not compiled, not run on hardware
//
#include <hip/hip_runtime.h>

// ---- problem dims ----
#define BATCH   2048
#define MSLOTS  64
#define DDIM    512
#define LDSROW  520          // padded LDS row stride (ushorts): 1040B, 16B-aligned

typedef __bf16 bf16_t;
typedef bf16_t v16bf __attribute__((ext_vector_type(16)));
typedef float  v8f   __attribute__((ext_vector_type(8)));
typedef unsigned int v4u __attribute__((ext_vector_type(4)));
typedef unsigned short u16;

union ABreg { v16bf v; v4u q[2]; };

// ---- workspace layout (bytes) ----
// Packed B-operand weights: layout [ct][kt][lane][16 bf16] -> each 32x16 WMMA
// B tile is one contiguous 1KB block (fully coalesced wave fetch).
#define W1CAT_OFF   0u                              // 5 * 512*1024 bf16 (packed, K=1024)
#define W1CAT_BYTES (5u * 512u * 1024u * 2u)        // 5,242,880
#define WSUB_OFF    (W1CAT_OFF + W1CAT_BYTES)       // 512*512 bf16 (packed, K=512)
#define WSUB_BYTES  (512u * 512u * 2u)              // 524,288
#define XB_OFF      (WSUB_OFF + WSUB_BYTES)         // 2048*512 bf16 (row-major)
#define XB_BYTES    (2048u * 512u * 2u)             // 2,097,152
#define XH_OFF      (XB_OFF + XB_BYTES)             // 2048*2560 f32
// total ~28.8 MB

__device__ __forceinline__ u16 f2bf(float x) {
    union { float f; unsigned u; } v; v.f = x;
    unsigned r = (v.u + 0x7FFFu + ((v.u >> 16) & 1u)) >> 16;   // RNE
    return (u16)r;
}
__device__ __forceinline__ float bf2f(u16 h) {
    union { unsigned u; float f; } v; v.u = ((unsigned)h) << 16;
    return v.f;
}
__device__ __forceinline__ float sigmoidf(float x) {
    return 1.0f / (1.0f + __expf(-x));
}

__device__ __forceinline__ v8f wmma_bf16(v16bf a, v16bf b, v8f c) {
    return __builtin_amdgcn_wmma_f32_16x16x32_bf16(
        /*neg_a=*/false, a, /*neg_b=*/false, b,
        /*c_mod=*/(short)0, c, /*reuse_a=*/false, /*reuse_b=*/false);
}

// A operand (16x32 bf16, M x K) from row-major LDS (stride LDSROW ushorts).
// lanes 0-15: K = k0+0..7 / k0+16..23 ; lanes 16-31: K = k0+8..15 / k0+24..31
__device__ __forceinline__ v16bf ldA_lds(const u16* base, int rowBase, int k0, int lane) {
    int row = rowBase + (lane & 15);
    int kof = k0 + ((lane & 16) ? 8 : 0);
    const u16* p = base + row * LDSROW + kof;
    ABreg a;
    a.q[0] = *(const v4u*)p;
    a.q[1] = *(const v4u*)(p + 16);
    return a.v;
}

// B operand (32x16 bf16, K x N) from PACKED weights: tile (ct, kt) is a
// contiguous 1KB block; each lane reads its own contiguous 32 bytes.
__device__ __forceinline__ const u16* pkB_addr(const u16* pk, int ct, int kt, int KT, int lane) {
    return pk + (((((size_t)ct * KT) + kt) * 32 + lane) << 4);
}
__device__ __forceinline__ v16bf ldB_pk(const u16* pk, int ct, int kt, int KT, int lane) {
    const u16* p = pkB_addr(pk, ct, kt, KT, lane);
    ABreg b;
    b.q[0] = *(const v4u*)p;
    b.q[1] = *(const v4u*)(p + 8);
    return b.v;
}

// ---------------- kernel 1a: f32 -> bf16 (row-major, for x) ----------------
__global__ void cvt_bf16(const float* __restrict__ s, u16* __restrict__ d, int n) {
    for (int i = blockIdx.x * blockDim.x + threadIdx.x; i < n; i += gridDim.x * blockDim.x)
        d[i] = f2bf(s[i]);
}

// ---------------- kernel 1b: f32 weight [512, K] -> packed bf16 B-tiles ----------------
// dst index i: e = i&15, lane = (i>>4)&31, kt = (i>>9) % KT, ct = (i>>9) / KT
// src element: row = ct*16 + (lane&15), k = kt*32 + ((lane&16)?8:0) + (e<8 ? e : e+8)
__global__ void pack_b_bf16(const float* __restrict__ W, u16* __restrict__ dst, int K) {
    const int KT = K >> 5;
    const int total = 512 * K;
    for (int i = blockIdx.x * blockDim.x + threadIdx.x; i < total; i += gridDim.x * blockDim.x) {
        int e = i & 15;
        int lane = (i >> 4) & 31;
        int rem = i >> 9;
        int kt = rem % KT;
        int ct = rem / KT;
        int row = ct * 16 + (lane & 15);
        int k = kt * 32 + ((lane & 16) ? 8 : 0) + (e < 8 ? e : e + 8);
        dst[i] = f2bf(W[(size_t)row * K + k]);
    }
}

// ---------------- kernel 2: XH[b, g*512+d] = x[b] @ W1g[:,512:]^T ----------------
// grid (128 row-tiles of B, 160 col-tiles of 5*512), 1 wave per block
__global__ __launch_bounds__(32) void xh_kernel(const u16* __restrict__ xb,
                                                const u16* __restrict__ w1cat,
                                                float* __restrict__ xh) {
    int lane = threadIdx.x;
    int bt = blockIdx.x;          // batch row tile
    int ct = blockIdx.y;          // column tile over 5*512
    int n = lane & 15;
    int hi = lane & 16;
    int g = ct >> 5;              // gate id (32 col tiles per gate)
    int ctg = ct & 31;            // col tile within gate
    const u16* wg = w1cat + (size_t)g * 512 * 1024;   // packed, KT=32
    const u16* arow = xb + (size_t)(bt * 16 + n) * 512;

    v8f acc = {};
#pragma unroll
    for (int kk = 0; kk < 16; ++kk) {
        ABreg A;
        const u16* pa = arow + kk * 32 + (hi ? 8 : 0);
        A.q[0] = *(const v4u*)pa;
        A.q[1] = *(const v4u*)(pa + 16);
        // x-half of W1: K offsets 512.. -> packed kt = 16 + kk
        v16bf B = ldB_pk(wg, ctg, 16 + kk, 32, lane);
        acc = wmma_bf16(A.v, B, acc);
    }
    int rbase = hi ? 8 : 0;
    float* orow = xh + (size_t)(bt * 16) * 2560 + (size_t)ct * 16 + n;
#pragma unroll
    for (int r = 0; r < 8; ++r)
        orow[(size_t)(r + rbase) * 2560] = acc[r];
}

// ---------------- kernel 3: main fused module, one workgroup per batch ----------------
__global__ __launch_bounds__(256) void memmod_kernel(
    const float* __restrict__ sub,
    const float* __restrict__ b1r, const float* __restrict__ W2r,
    const float* __restrict__ b1w, const float* __restrict__ W2w,
    const float* __restrict__ bfv, const float* __restrict__ biv, const float* __restrict__ bcv,
    const float* __restrict__ bsubv, const float* __restrict__ Wfin, const float* __restrict__ bfin,
    const u16* __restrict__ w1cat, const u16* __restrict__ wsub_bf, const float* __restrict__ xh,
    float* __restrict__ out)
{
    __shared__ __attribute__((aligned(16))) u16 subb[MSLOTS * LDSROW];  // sub_memory bf16
    __shared__ __attribute__((aligned(16))) u16 memb[MSLOTS * LDSROW];  // gated mem bf16
    __shared__ float sr[64], sw[64], simr[64], simw[64], subvec[512];

    const int tid = threadIdx.x;
    const int b = blockIdx.x;
    const float* subB = sub + (size_t)b * MSLOTS * DDIM;
    const float* xhb = xh + (size_t)b * 2560;
    float* outB = out + (size_t)b * 65 * DDIM;

    // stage sub_memory[b] into LDS as bf16 (coalesced)
    for (int i = tid; i < MSLOTS * DDIM; i += 256) {
        int m = i >> 9, d = i & 511;
        subb[m * LDSROW + d] = f2bf(subB[i]);
    }
    if (tid < 64) { sr[tid] = 0.0f; sw[tid] = 0.0f; }
    __syncthreads();

    const int wv = tid >> 5, lane = tid & 31;
    const int rt = wv & 3, cgrp = wv >> 2;
    const int n = lane & 15, hi = lane & 16;
    const int rbase = rt * 16;
    const int rowoff = hi ? 8 : 0;

    // ---- Phase 1: addressing gates r/w: h = relu(sub@W1^T + x-part + b1); s += h . W2
    for (int g = 0; g < 2; ++g) {
        const u16* wg = w1cat + (size_t)g * 512 * 1024;   // packed, KT=32 (sub-half: kt 0..15)
        const float* b1 = g ? b1w : b1r;
        const float* w2 = g ? W2w : W2r;
        float* sdst = g ? sw : sr;
        for (int jb = 0; jb < 4; ++jb) {                  // 4 col tiles per A load
            int ct0 = cgrp * 16 + jb * 4;
            v8f acc[4] = {{}, {}, {}, {}};
            if (jb < 3)  // prefetch next block of packed B (gfx1250 global_prefetch)
                __builtin_prefetch(pkB_addr(wg, ct0 + 4, 0, 32, lane), 0, 3);
#pragma unroll
            for (int kt = 0; kt < 16; ++kt) {
                v16bf A = ldA_lds(subb, rbase, kt * 32, lane);
#pragma unroll
                for (int t = 0; t < 4; ++t)
                    acc[t] = wmma_bf16(A, ldB_pk(wg, ct0 + t, kt, 32, lane), acc[t]);
            }
#pragma unroll
            for (int t = 0; t < 4; ++t) {
                int col = (ct0 + t) * 16 + n;
                float add = xhb[g * 512 + col] + b1[col];
                float w2v = w2[col];
#pragma unroll
                for (int r = 0; r < 8; ++r) {
                    float h = acc[t][r] + add;
                    h = h > 0.0f ? h : 0.0f;
                    atomicAdd(&sdst[rbase + rowoff + r], h * w2v);  // b2 dropped: softmax-invariant
                }
            }
        }
    }
    __syncthreads();

    // ---- Phase 2: softmax over slots; sub_vec; read_vec = tanh(sub_vec @ Wfin^T + bfin)
    if (tid < 64) {
        float mx = -1e30f;
        for (int j = 0; j < 64; ++j) mx = fmaxf(mx, sr[j]);
        float s = 0.0f;
        for (int j = 0; j < 64; ++j) s += __expf(sr[j] - mx);
        simr[tid] = __expf(sr[tid] - mx) / s;
    } else if (tid < 128) {
        int t = tid - 64;
        float mx = -1e30f;
        for (int j = 0; j < 64; ++j) mx = fmaxf(mx, sw[j]);
        float s = 0.0f;
        for (int j = 0; j < 64; ++j) s += __expf(sw[j] - mx);
        simw[t] = __expf(sw[t] - mx) / s;
    }
    __syncthreads();
    for (int d = tid; d < 512; d += 256) {
        float a0 = 0.0f;
        for (int m = 0; m < 64; ++m) a0 += bf2f(subb[m * LDSROW + d]) * simr[m];
        subvec[d] = a0;
    }
    __syncthreads();
    for (int e = tid; e < 512; e += 256) {
        const float* wrow = Wfin + (size_t)e * 512;
        float a0 = bfin[e];
        for (int d = 0; d < 512; ++d) a0 += subvec[d] * wrow[d];
        outB[e] = tanhf(a0);
    }
    __syncthreads();

    // ---- Phase 3: f/i/c (A reused across 3 gates x 2 tiles) + gated update -> memb
    {
        const u16* wf = w1cat + 2 * (size_t)512 * 1024;
        const u16* wi = w1cat + 3 * (size_t)512 * 1024;
        const u16* wc = w1cat + 4 * (size_t)512 * 1024;
        for (int jb = 0; jb < 8; ++jb) {                  // 2 col tiles per A load
            int ct0 = cgrp * 16 + jb * 2;
            v8f af[2] = {{}, {}}, ai[2] = {{}, {}}, ac[2] = {{}, {}};
            if (jb < 7) {
                __builtin_prefetch(pkB_addr(wf, ct0 + 2, 0, 32, lane), 0, 3);
                __builtin_prefetch(pkB_addr(wi, ct0 + 2, 0, 32, lane), 0, 3);
                __builtin_prefetch(pkB_addr(wc, ct0 + 2, 0, 32, lane), 0, 3);
            }
#pragma unroll
            for (int kt = 0; kt < 16; ++kt) {
                v16bf A = ldA_lds(subb, rbase, kt * 32, lane);
#pragma unroll
                for (int t = 0; t < 2; ++t) {
                    af[t] = wmma_bf16(A, ldB_pk(wf, ct0 + t, kt, 32, lane), af[t]);
                    ai[t] = wmma_bf16(A, ldB_pk(wi, ct0 + t, kt, 32, lane), ai[t]);
                    ac[t] = wmma_bf16(A, ldB_pk(wc, ct0 + t, kt, 32, lane), ac[t]);
                }
            }
#pragma unroll
            for (int t = 0; t < 2; ++t) {
                int col = (ct0 + t) * 16 + n;
                float addf = xhb[2 * 512 + col] + bfv[col];
                float addi = xhb[3 * 512 + col] + biv[col];
                float addc = xhb[4 * 512 + col] + bcv[col];
#pragma unroll
                for (int r = 0; r < 8; ++r) {
                    int m = rbase + rowoff + r;
                    float fg = sigmoidf(af[t][r] + addf);
                    float ig = sigmoidf(ai[t][r] + addi);
                    float cg = tanhf(ac[t][r] + addc);
                    float sm = simw[m];
                    float sv = bf2f(subb[m * LDSROW + col]);
                    float mv = (1.0f - sm * fg) * sv + ig * cg * sm;
                    memb[m * LDSROW + col] = f2bf(mv);
                }
            }
        }
    }
    __syncthreads();

    // ---- Phase 4: new_mem = tanh(mem @ Wsub^T + bsub), packed Wsub KT=16
    for (int jb = 0; jb < 4; ++jb) {
        int ct0 = cgrp * 16 + jb * 4;
        v8f acc[4] = {{}, {}, {}, {}};
        if (jb < 3)
            __builtin_prefetch(pkB_addr(wsub_bf, ct0 + 4, 0, 16, lane), 0, 3);
#pragma unroll
        for (int kt = 0; kt < 16; ++kt) {
            v16bf A = ldA_lds(memb, rbase, kt * 32, lane);
#pragma unroll
            for (int t = 0; t < 4; ++t)
                acc[t] = wmma_bf16(A, ldB_pk(wsub_bf, ct0 + t, kt, 16, lane), acc[t]);
        }
#pragma unroll
        for (int t = 0; t < 4; ++t) {
            int col = (ct0 + t) * 16 + n;
            float bb = bsubv[col];
#pragma unroll
            for (int r = 0; r < 8; ++r) {
                int m = rbase + rowoff + r;
                outB[(size_t)(1 + m) * DDIM + col] = tanhf(acc[t][r] + bb);
            }
        }
    }
}

extern "C" void kernel_launch(void* const* d_in, const int* in_sizes, int n_in,
                              void* d_out, int out_size, void* d_ws, size_t ws_size,
                              hipStream_t stream) {
    (void)in_sizes; (void)n_in; (void)out_size; (void)ws_size;
    char* ws = (char*)d_ws;
    u16* w1cat   = (u16*)(ws + W1CAT_OFF);
    u16* wsub_bf = (u16*)(ws + WSUB_OFF);
    u16* xb      = (u16*)(ws + XB_OFF);
    float* xhbuf = (float*)(ws + XH_OFF);

    const float* x     = (const float*)d_in[0];
    const float* subm  = (const float*)d_in[1];
    const float* W1r   = (const float*)d_in[2];
    const float* b1r   = (const float*)d_in[3];
    const float* W2r   = (const float*)d_in[4];
    const float* W1w   = (const float*)d_in[6];
    const float* b1w   = (const float*)d_in[7];
    const float* W2w   = (const float*)d_in[8];
    const float* Wf    = (const float*)d_in[10];
    const float* bfv   = (const float*)d_in[11];
    const float* Wi    = (const float*)d_in[12];
    const float* biv   = (const float*)d_in[13];
    const float* Wc    = (const float*)d_in[14];
    const float* bcv   = (const float*)d_in[15];
    const float* Wsub  = (const float*)d_in[16];
    const float* bsubv = (const float*)d_in[17];
    const float* Wfin  = (const float*)d_in[18];
    const float* bfinv = (const float*)d_in[19];

    const size_t WN = 512 * 1024;
    pack_b_bf16<<<1024, 256, 0, stream>>>(W1r, w1cat + 0 * WN, 1024);
    pack_b_bf16<<<1024, 256, 0, stream>>>(W1w, w1cat + 1 * WN, 1024);
    pack_b_bf16<<<1024, 256, 0, stream>>>(Wf,  w1cat + 2 * WN, 1024);
    pack_b_bf16<<<1024, 256, 0, stream>>>(Wi,  w1cat + 3 * WN, 1024);
    pack_b_bf16<<<1024, 256, 0, stream>>>(Wc,  w1cat + 4 * WN, 1024);
    pack_b_bf16<<<512, 256, 0, stream>>>(Wsub, wsub_bf, 512);
    cvt_bf16<<<1024, 256, 0, stream>>>(x, xb, 2048 * 512);

    dim3 gxh(128, 160);
    xh_kernel<<<gxh, 32, 0, stream>>>(xb, w1cat, xhbuf);

    memmod_kernel<<<BATCH, 256, 0, stream>>>(
        subm, b1r, W2r, b1w, W2w, bfv, biv, bcv,
        bsubv, Wfin, bfinv, w1cat, wsub_bf, xhbuf, (float*)d_out);
}
